// HNetAttention_43619687858583
// MI455X (gfx1250) — matrix operand order repaired and checked by reading, not compiled, not run on hardware
//
#include <hip/hip_runtime.h>

// ---------------------------------------------------------------------------
// CDNA5 (gfx1250, wave32) attention layer.
//   convert   : one-time fp32 -> bf16 pack of X and all weights
//   qkv_rope  : Xb @ {Wq,Wk,Wv}b^T (v_wmma_f32_16x16x32_bf16) + RoPE
//               weights staged per-block in double-buffered LDS (8x L2 reuse)
//   attn      : causal flash-attention; K/V blocks staged in LDS per block
//   oproj     : attn_out @ Wo^T -> fp32, Wo staged in LDS
// Staging uses gfx1250 async global->LDS (global_load_async_to_lds_b128 +
// s_wait_asynccnt) when available, else plain load+ds_store fallback.
// ---------------------------------------------------------------------------

typedef __attribute__((ext_vector_type(16))) __bf16 v16bf;
typedef __attribute__((ext_vector_type(8)))  float  v8f;
typedef __attribute__((ext_vector_type(4)))  int    v4i;

#define HID   768
#define SEQ   2048
#define NHEAD 12
#define HDIM  64

union f8u  { v8f v; float f[8]; };
union bfr  { v16bf v; __bf16 e[16]; uint4 u[2]; };

#if defined(__has_builtin)
#if __has_builtin(__builtin_amdgcn_global_load_async_to_lds_b128) && \
    __has_builtin(__builtin_amdgcn_s_wait_asynccnt)
#define USE_ASYNC_LDS 1
#endif
#endif
#ifndef USE_ASYNC_LDS
#define USE_ASYNC_LDS 0
#endif

#if USE_ASYNC_LDS
typedef __attribute__((address_space(1))) v4i gv4i;   // global
typedef __attribute__((address_space(3))) v4i lv4i;   // LDS
#endif

// copy 16 bytes global -> LDS (async DMA path on gfx1250, sync fallback)
__device__ __forceinline__ void cp16(__bf16* l, const __bf16* g) {
#if USE_ASYNC_LDS
  __builtin_amdgcn_global_load_async_to_lds_b128((gv4i*)(void*)g, (lv4i*)l, 0, 0);
#else
  *(uint4*)l = *(const uint4*)g;
#endif
}
__device__ __forceinline__ void cp_wait() {
#if USE_ASYNC_LDS
  __builtin_amdgcn_s_wait_asynccnt(0);
#endif
}

__device__ __forceinline__ unsigned rne_hi16(float f) {
  unsigned u = __builtin_bit_cast(unsigned, f);
  u += 0x7FFFu + ((u >> 16) & 1u);             // round-to-nearest-even
  return u >> 16;
}
__device__ __forceinline__ __bf16 f2bf(float f) {
  unsigned short h = (unsigned short)rne_hi16(f);
  return __builtin_bit_cast(__bf16, h);
}
__device__ __forceinline__ unsigned pack2(float lo, float hi) {
  return rne_hi16(lo) | (rne_hi16(hi) << 16);
}

__device__ __forceinline__ v8f wmma_bf16(v16bf a, v16bf b, v8f c) {
  return __builtin_amdgcn_wmma_f32_16x16x32_bf16(false, a, false, b, (short)0, c, false, false);
}

// A fragment (16x32 bf16): lane row = lane%16; elems j -> K = 2*h8 + j (16 contiguous)
__device__ __forceinline__ v16bf load_a_bf16(const __bf16* p, int h8) {
  bfr f;
  f.u[0] = *(const uint4*)(p + 2 * h8);
  f.u[1] = *(const uint4*)(p + 2 * h8 + 8);
  return f.v;
}
// B fragment (32x16 bf16): lane col = lane%16; elems j -> K = (lane/16)*16 + j (16 contiguous)
__device__ __forceinline__ v16bf load_b16(const __bf16* p) {
  bfr f;
  f.u[0] = *(const uint4*)(p);
  f.u[1] = *(const uint4*)(p + 8);
  return f.v;
}

// ---------------------------------------------------------------------------
// Kernel 0: fp32 -> bf16 bulk convert
// ---------------------------------------------------------------------------
__global__ __launch_bounds__(256) void convert_kernel(const float* __restrict__ src,
                                                      __bf16* __restrict__ dst, int n8) {
  for (int i = blockIdx.x * blockDim.x + threadIdx.x; i < n8; i += gridDim.x * blockDim.x) {
    const float4 a = ((const float4*)src)[2 * i];
    const float4 b = ((const float4*)src)[2 * i + 1];
    uint4 o;
    o.x = pack2(a.x, a.y); o.y = pack2(a.z, a.w);
    o.z = pack2(b.x, b.y); o.w = pack2(b.z, b.w);
    ((uint4*)dst)[i] = o;
  }
}

// ---------------------------------------------------------------------------
// Kernel 1: QKV + RoPE.  Block = 8 m-tiles x 1 head; W k-blocks shared via LDS.
// ---------------------------------------------------------------------------
__global__ __launch_bounds__(256) void qkv_rope_kernel(
    const __bf16* __restrict__ X, const __bf16* __restrict__ Wq,
    const __bf16* __restrict__ Wk, const __bf16* __restrict__ Wv,
    __bf16* __restrict__ Q, __bf16* __restrict__ K, __bf16* __restrict__ Vt) {
  __shared__ __bf16 wlds[2][3][64][40];            // [buf][mat][row n][k 32 + pad]
  const int tid = threadIdx.x;
  const int lane = tid & 31, wv = tid >> 5;
  const int hh = blockIdx.x >> 6;                  // 0..11 (same head across 64 blocks)
  const int mg = blockIdx.x & 63;
  const int m0 = (mg * 8 + wv) * 16;
  const int ln = lane & 15, hb = lane >> 4, h8 = hb * 8, h16 = hb * 16;
  const int crow = tid >> 2, cpart = (tid & 3) * 8;  // 64 rows x 4 chunks of 8 bf16

  f8u accQ[4], accK[4], accV[4];
#pragma unroll
  for (int t = 0; t < 4; ++t)
#pragma unroll
    for (int j = 0; j < 8; ++j) { accQ[t].f[j] = 0.f; accK[t].f[j] = 0.f; accV[t].f[j] = 0.f; }

  const __bf16* xrow = X + (size_t)(m0 + ln) * HID;
  const size_t woff = (size_t)(hh * HDIM + crow) * HID + cpart;
  for (int kb = 0; kb < HID; kb += 32) {
    const int p = (kb >> 5) & 1;
    cp16(&wlds[p][0][crow][cpart], Wq + woff + kb);
    cp16(&wlds[p][1][crow][cpart], Wk + woff + kb);
    cp16(&wlds[p][2][crow][cpart], Wv + woff + kb);
    cp_wait();
    __syncthreads();                               // double buffer: one barrier/step
    v16bf a = load_a_bf16(xrow + kb, h8);
    __builtin_prefetch(xrow + kb + 32, 0, 3);
#pragma unroll
    for (int t = 0; t < 4; ++t) {
      const int rn = t * 16 + ln;
      accQ[t].v = wmma_bf16(a, load_b16(&wlds[p][0][rn][h16]), accQ[t].v);
      accK[t].v = wmma_bf16(a, load_b16(&wlds[p][1][rn][h16]), accK[t].v);
      accV[t].v = wmma_bf16(a, load_b16(&wlds[p][2][rn][h16]), accV[t].v);
    }
  }

  // RoPE: pair (d, d+32) = (tile t, t+2), same lane.  C layout: row = j + 8*hb.
  const int s0 = m0 & (SEQ - 1);
#pragma unroll
  for (int t = 0; t < 2; ++t) {
    const int d = t * 16 + ln;
    const float invf = powf(10000.0f, -(float)d * (1.0f / 32.0f));
#pragma unroll
    for (int j = 0; j < 8; ++j) {
      const float ang = (float)(s0 + j + h8) * invf;
      float sn, cs; sincosf(ang, &sn, &cs);
      float q1 = accQ[t].f[j], q2 = accQ[t + 2].f[j];
      accQ[t].f[j]     = q1 * cs - q2 * sn;
      accQ[t + 2].f[j] = q2 * cs + q1 * sn;
      float k1 = accK[t].f[j], k2 = accK[t + 2].f[j];
      accK[t].f[j]     = k1 * cs - k2 * sn;
      accK[t + 2].f[j] = k2 * cs + k1 * sn;
    }
  }

  const int bh = (m0 >> 11) * NHEAD + hh;
#pragma unroll
  for (int t = 0; t < 4; ++t)
#pragma unroll
    for (int j = 0; j < 8; ++j) {
      const int srow = (m0 & (SEQ - 1)) + j + h8;
      const int d = t * 16 + ln;
      Q[((size_t)bh * SEQ + srow) * HDIM + d]  = f2bf(accQ[t].f[j]);
      K[((size_t)bh * SEQ + srow) * HDIM + d]  = f2bf(accK[t].f[j]);
      Vt[((size_t)bh * HDIM + d) * SEQ + srow] = f2bf(accV[t].f[j]);   // transposed
    }
}

// ---------------------------------------------------------------------------
// Kernel 2: causal flash-attention.  Block = 8 query-tiles x 1 head;
// K/V 32-key blocks staged in LDS and shared by all 8 waves.
// ---------------------------------------------------------------------------
__global__ __launch_bounds__(256) void attn_kernel(
    const __bf16* __restrict__ Q, const __bf16* __restrict__ K,
    const __bf16* __restrict__ Vt, __bf16* __restrict__ AO) {
  __shared__ __bf16 klds[2][32][72];               // [buf][key][d 64 + pad]
  __shared__ __bf16 vlds[2][64][40];               // [buf][d][key 32 + pad]
  __shared__ __bf16 plds[8][16][40];               // per-wave P tile transpose
  const int tid = threadIdx.x;
  const int lane = tid & 31, wv = tid >> 5;
  const int b  = blockIdx.x / (NHEAD * 16);
  const int r  = blockIdx.x % (NHEAD * 16);
  const int hh = r >> 4;
  const int qg = r & 15;
  const int q0 = (qg * 8 + wv) * 16;
  const int bh = b * NHEAD + hh;
  const int kend = (qg + 1) * 128;                 // block-uniform causal bound
  const int ln = lane & 15, hb = lane >> 4, h8 = hb * 8, h16 = hb * 16;
  const int krow = tid >> 3, kpart = (tid & 7) * 8;   // K stage: 32 rows x 8 chunks
  const int vrow = tid >> 2, vpart = (tid & 3) * 8;   // V stage: 64 rows x 4 chunks

  const __bf16* qrow = Q + ((size_t)bh * SEQ + q0 + ln) * HDIM;
  const v16bf aq0 = load_a_bf16(qrow, h8);         // d = 0..31
  const v16bf aq1 = load_a_bf16(qrow + 32, h8);    // d = 32..63

  f8u accO[4];
  float mrun[8], lrun[8];
#pragma unroll
  for (int j = 0; j < 8; ++j) { mrun[j] = -1e30f; lrun[j] = 0.f; }
#pragma unroll
  for (int t = 0; t < 4; ++t)
#pragma unroll
    for (int j = 0; j < 8; ++j) accO[t].f[j] = 0.f;

  for (int kb = 0; kb < kend; kb += 32) {
    const int p = (kb >> 5) & 1;
    {
      int key = kb + krow; if (key > SEQ - 1) key = SEQ - 1;
      cp16(&klds[p][krow][kpart], K + ((size_t)bh * SEQ + key) * HDIM + kpart);
      int kc = kb + vpart; if (kc > SEQ - 8) kc = SEQ - 8;
      cp16(&vlds[p][vrow][vpart], Vt + ((size_t)bh * HDIM + vrow) * SEQ + kc);
    }
    cp_wait();
    __syncthreads();                               // double buffer: one barrier/step
    if (kb < q0 + 16) {                            // wave-uniform: EXEC stays all-ones
      // ---- scores S = Q K^T (16 x 32) ----
      f8u sc[2];
#pragma unroll
      for (int kt = 0; kt < 2; ++kt) {
#pragma unroll
        for (int j = 0; j < 8; ++j) sc[kt].f[j] = 0.f;
        const int rn = kt * 16 + ln;
        sc[kt].v = wmma_bf16(aq0, load_b16(&klds[p][rn][h16]), sc[kt].v);
        sc[kt].v = wmma_bf16(aq1, load_b16(&klds[p][rn][32 + h16]), sc[kt].v);
      }
      // ---- mask + online softmax (a row lives across a 16-lane half) ----
      float pc[2][8];
#pragma unroll
      for (int j = 0; j < 8; ++j) {
        const int qr = q0 + j + h8;
        float s0v = sc[0].f[j] * 0.125f; if (kb + ln      > qr) s0v = -1e30f;
        float s1v = sc[1].f[j] * 0.125f; if (kb + 16 + ln > qr) s1v = -1e30f;
        float mx = fmaxf(s0v, s1v);
#pragma unroll
        for (int o = 1; o < 16; o <<= 1) mx = fmaxf(mx, __shfl_xor(mx, o, 16));
        const float mnew = fmaxf(mrun[j], mx);
        const float al = expf(mrun[j] - mnew);
        mrun[j] = mnew;
        s0v = expf(s0v - mnew); s1v = expf(s1v - mnew);
        float rs = s0v + s1v;
#pragma unroll
        for (int o = 1; o < 16; o <<= 1) rs += __shfl_xor(rs, o, 16);
        lrun[j] = lrun[j] * al + rs;
#pragma unroll
        for (int t = 0; t < 4; ++t) accO[t].f[j] *= al;
        pc[0][j] = s0v; pc[1][j] = s1v;
      }
      // ---- P: C-layout -> A-layout via LDS (same-wave DS ops in order) ----
#pragma unroll
      for (int kt = 0; kt < 2; ++kt)
#pragma unroll
        for (int j = 0; j < 8; ++j)
          plds[wv][j + h8][kt * 16 + ln] = f2bf(pc[kt][j]);
      bfr ap;
      const __bf16* prow = &plds[wv][ln][0];
      ap.u[0] = *(const uint4*)(prow + 2 * h8);
      ap.u[1] = *(const uint4*)(prow + 2 * h8 + 8);
      // ---- O += P V ----
#pragma unroll
      for (int t = 0; t < 4; ++t)
        accO[t].v = wmma_bf16(ap.v, load_b16(&vlds[p][t * 16 + ln][h16]), accO[t].v);
    }
  }
  // ---- epilogue ----
#pragma unroll
  for (int t = 0; t < 4; ++t)
#pragma unroll
    for (int j = 0; j < 8; ++j) {
      const size_t m = (size_t)b * SEQ + q0 + j + h8;
      AO[m * HID + hh * HDIM + t * 16 + ln] = f2bf(accO[t].f[j] / lrun[j]);
    }
}

// ---------------------------------------------------------------------------
// Kernel 3: output projection.  Block = 8 m-tiles x 64 features; Wo via LDS.
// ---------------------------------------------------------------------------
__global__ __launch_bounds__(256) void oproj_kernel(
    const __bf16* __restrict__ AO, const __bf16* __restrict__ Wo,
    float* __restrict__ Out) {
  __shared__ __bf16 wlds[2][64][40];
  const int tid = threadIdx.x;
  const int lane = tid & 31, wv = tid >> 5;
  const int ns = blockIdx.x >> 6;
  const int mg = blockIdx.x & 63;
  const int m0 = (mg * 8 + wv) * 16, n0 = ns * HDIM;
  const int ln = lane & 15, hb = lane >> 4, h8 = hb * 8, h16 = hb * 16;
  const int crow = tid >> 2, cpart = (tid & 3) * 8;

  f8u acc[4];
#pragma unroll
  for (int t = 0; t < 4; ++t)
#pragma unroll
    for (int j = 0; j < 8; ++j) acc[t].f[j] = 0.f;

  const __bf16* arow = AO + (size_t)(m0 + ln) * HID;
  const size_t woff = (size_t)(n0 + crow) * HID + cpart;
  for (int kb = 0; kb < HID; kb += 32) {
    const int p = (kb >> 5) & 1;
    cp16(&wlds[p][crow][cpart], Wo + woff + kb);
    cp_wait();
    __syncthreads();
    v16bf a = load_a_bf16(arow + kb, h8);
    __builtin_prefetch(arow + kb + 32, 0, 3);
#pragma unroll
    for (int t = 0; t < 4; ++t)
      acc[t].v = wmma_bf16(a, load_b16(&wlds[p][t * 16 + ln][h16]), acc[t].v);
  }
#pragma unroll
  for (int t = 0; t < 4; ++t)
#pragma unroll
    for (int j = 0; j < 8; ++j)
      Out[(size_t)(m0 + j + h8) * HID + n0 + t * 16 + ln] = acc[t].f[j];
}

// ---------------------------------------------------------------------------
extern "C" void kernel_launch(void* const* d_in, const int* in_sizes, int n_in,
                              void* d_out, int out_size, void* d_ws, size_t ws_size,
                              hipStream_t stream) {
  const float* X  = (const float*)d_in[0];
  const float* Wq = (const float*)d_in[1];
  const float* Wk = (const float*)d_in[2];
  const float* Wv = (const float*)d_in[3];
  const float* Wo = (const float*)d_in[4];
  float* out = (float*)d_out;

  const size_t NTOK = (size_t)4 * SEQ;            // 8192 tokens
  const size_t ELTS = NTOK * HID;                 // 6291456 per activation buffer
  const size_t WELT = (size_t)HID * HID;          // 589824 per weight matrix
  __bf16* Q   = (__bf16*)d_ws;
  __bf16* K   = Q   + ELTS;
  __bf16* Vt  = K   + ELTS;
  __bf16* AO  = Vt  + ELTS;
  __bf16* Xb  = AO  + ELTS;
  __bf16* Wqb = Xb  + ELTS;
  __bf16* Wkb = Wqb + WELT;
  __bf16* Wvb = Wkb + WELT;
  __bf16* Wob = Wvb + WELT;

  convert_kernel<<<1024, 256, 0, stream>>>(X,  Xb,  (int)(ELTS / 8));
  convert_kernel<<<288,  256, 0, stream>>>(Wq, Wqb, (int)(WELT / 8));
  convert_kernel<<<288,  256, 0, stream>>>(Wk, Wkb, (int)(WELT / 8));
  convert_kernel<<<288,  256, 0, stream>>>(Wv, Wvb, (int)(WELT / 8));
  convert_kernel<<<288,  256, 0, stream>>>(Wo, Wob, (int)(WELT / 8));

  const int blocks = 768;                         // 8 waves (256 thr) per block
  qkv_rope_kernel<<<blocks, 256, 0, stream>>>(Xb, Wqb, Wkb, Wvb, Q, K, Vt);
  attn_kernel<<<blocks, 256, 0, stream>>>(Q, K, Vt, AO);
  oproj_kernel<<<blocks, 256, 0, stream>>>(AO, Wob, out);
}
